// SWE_Pooling_52175262712328
// MI455X (gfx1250) — compile-verified, matrix operand order checked by reading, not run on hardware
//
#include <hip/hip_runtime.h>
#include <hip/hip_bf16.h>

// ---------------------------------------------------------------------------
// SWE pooling for MI455X (gfx1250, wave32, WMMA).
//   B=32, N=2048, D=512, L=512, M=256
//   out[b, l*M + m] = Rslices[m,l] - interp(sort_n(X[b]·w[l]))[Rind[m,l]]
// ---------------------------------------------------------------------------

typedef __attribute__((ext_vector_type(16))) __bf16 v16bf;
typedef __attribute__((ext_vector_type(8)))  __bf16 v8bf;
typedef __attribute__((ext_vector_type(8)))  float  v8f;

#define D_IN  512
#define NPTS  2048
#define BATCH 32
#define LSL   512
#define MREF  256

// ---------------------------------------------------------------------------
// 1. Row-normalize theta_v -> w  (weight_norm with g == 1)
// ---------------------------------------------------------------------------
__global__ void __launch_bounds__(256)
normalize_rows_kernel(const float* __restrict__ theta, float* __restrict__ w) {
  __shared__ float red[256];
  const int l = blockIdx.x;
  const int t = threadIdx.x;
  const float x0 = theta[l * D_IN + t];
  const float x1 = theta[l * D_IN + 256 + t];
  red[t] = x0 * x0 + x1 * x1;
  __syncthreads();
  for (int s = 128; s > 0; s >>= 1) {
    if (t < s) red[t] += red[t + s];
    __syncthreads();
  }
  const float inv = rsqrtf(red[0]);
  w[l * D_IN + t]       = x0 * inv;
  w[l * D_IN + 256 + t] = x1 * inv;
}

// ---------------------------------------------------------------------------
// 2. GEMM: XsT[b][l][n] = sum_d X[b*N+n][d] * w[l][d]   (bf16 WMMA, f32 acc)
//    Workgroup tile 128(M=b*N rows) x 128(L), 8 waves, each wave 32x64.
//    fp32 -> bf16 conversion happens while staging tiles into LDS.
// ---------------------------------------------------------------------------
__global__ void __launch_bounds__(256)
gemm_xw_kernel(const float* __restrict__ X, const float* __restrict__ w,
               float* __restrict__ XsT) {
  __shared__ __align__(16) __bf16 ldsA[128 * 32];
  __shared__ __align__(16) __bf16 ldsB[128 * 32];

  const int tid   = threadIdx.x;
  const int mBase = blockIdx.x * 128;   // over B*N = 65536
  const int lBase = blockIdx.y * 128;   // over L = 512
  const int lane  = tid & 31;
  const int wave  = tid >> 5;
  const int waveM = wave & 3;           // 4 waves along M
  const int waveN = wave >> 2;          // 2 waves along N(L)
  const int lr    = lane & 15;
  const int kc    = (lane >> 4);        // chunk select: 0 -> K{0..7,16..23}, 1 -> K{8..15,24..31}

  v8f acc[2][4];
  const v8f zero = {0.f, 0.f, 0.f, 0.f, 0.f, 0.f, 0.f, 0.f};
  #pragma unroll
  for (int i = 0; i < 2; ++i)
    #pragma unroll
    for (int j = 0; j < 4; ++j) acc[i][j] = zero;

  const int rowL = tid >> 1;            // 0..127  (tile row staged by this thread)
  const int colH = (tid & 1) * 16;      // 0 or 16 (16 fp32 = 64B half-row)

  for (int k0 = 0; k0 < D_IN; k0 += 32) {
    __syncthreads();
    // stage A (X rows) with fp32 -> bf16 convert
    {
      const float4* src = (const float4*)(X + (size_t)(mBase + rowL) * D_IN + k0 + colH);
      __bf16* dst = ldsA + rowL * 32 + colH;
      #pragma unroll
      for (int q = 0; q < 4; ++q) {
        const float4 v = src[q];
        dst[q * 4 + 0] = (__bf16)v.x;  dst[q * 4 + 1] = (__bf16)v.y;
        dst[q * 4 + 2] = (__bf16)v.z;  dst[q * 4 + 3] = (__bf16)v.w;
      }
      if (k0 + 32 < D_IN)  // global_prefetch_b8 of next K tile
        __builtin_prefetch(X + (size_t)(mBase + rowL) * D_IN + k0 + 32 + colH, 0, 0);
    }
    // stage B (w rows; B = w^T so B-column n == w row l, contiguous in k)
    {
      const float4* src = (const float4*)(w + (size_t)(lBase + rowL) * D_IN + k0 + colH);
      __bf16* dst = ldsB + rowL * 32 + colH;
      #pragma unroll
      for (int q = 0; q < 4; ++q) {
        const float4 v = src[q];
        dst[q * 4 + 0] = (__bf16)v.x;  dst[q * 4 + 1] = (__bf16)v.y;
        dst[q * 4 + 2] = (__bf16)v.z;  dst[q * 4 + 3] = (__bf16)v.w;
      }
    }
    __syncthreads();

    // fragments: ISA 16-bit A layout -> two aligned v8bf chunks per lane
    union Frag { v16bf v; v8bf h[2]; };
    Frag a[2], b[4];
    #pragma unroll
    for (int mt = 0; mt < 2; ++mt) {
      const int row = waveM * 32 + mt * 16 + lr;
      const v8bf* p = (const v8bf*)(ldsA + row * 32);
      a[mt].h[0] = p[kc];
      a[mt].h[1] = p[kc + 2];
    }
    #pragma unroll
    for (int nt = 0; nt < 4; ++nt) {
      const int col = waveN * 64 + nt * 16 + lr;
      const v8bf* p = (const v8bf*)(ldsB + col * 32);
      b[nt].h[0] = p[kc];
      b[nt].h[1] = p[kc + 2];
    }

    #pragma unroll
    for (int mt = 0; mt < 2; ++mt)
      #pragma unroll
      for (int nt = 0; nt < 4; ++nt)
        acc[mt][nt] = __builtin_amdgcn_wmma_f32_16x16x32_bf16(
            false, a[mt].v, false, b[nt].v, (short)0, acc[mt][nt], false, false);
  }

  // Epilogue: write transposed [b][l][n]. Per lane the 8 acc VGPRs are 8
  // consecutive n values -> two contiguous float4 stores.
  const int bb   = mBase >> 11;     // mBase / 2048 (tile never crosses a batch)
  const int nBlk = mBase & 2047;
  #pragma unroll
  for (int mt = 0; mt < 2; ++mt) {
    const int n0 = nBlk + waveM * 32 + mt * 16 + ((lane >> 4) << 3);
    #pragma unroll
    for (int nt = 0; nt < 4; ++nt) {
      const int l = lBase + waveN * 64 + nt * 16 + lr;
      float* dst = XsT + ((size_t)bb * LSL + l) * NPTS + n0;
      union { v8f v; float4 q[2]; } u;
      u.v = acc[mt][nt];
      ((float4*)dst)[0] = u.q[0];
      ((float4*)dst)[1] = u.q[1];
    }
  }
}

// ---------------------------------------------------------------------------
// 3. Rslices[m,l] = dot(ref_pts[m], w[l])   (tiny; one block per l)
// ---------------------------------------------------------------------------
__global__ void __launch_bounds__(256)
rslices_kernel(const float* __restrict__ ref, const float* __restrict__ w,
               float* __restrict__ Rs) {
  __shared__ float wl[D_IN];
  const int l = blockIdx.x;
  const int t = threadIdx.x;
  wl[t]       = w[l * D_IN + t];
  wl[t + 256] = w[l * D_IN + 256 + t];
  __syncthreads();
  float acc = 0.f;
  const float* rm = ref + (size_t)t * D_IN;   // thread t == ref point m
  for (int d = 0; d < D_IN; ++d) acc = fmaf(rm[d], wl[d], acc);
  Rs[t * LSL + l] = acc;
}

// ---------------------------------------------------------------------------
// 4. Rind[:,l] = argsort(Rslices[:,l])  (bitonic 256 key/index in LDS)
// ---------------------------------------------------------------------------
__global__ void __launch_bounds__(256)
argsort_kernel(const float* __restrict__ Rs, int* __restrict__ Rind) {
  __shared__ float key[256];
  __shared__ int   idx[256];
  const int l = blockIdx.x;
  const int t = threadIdx.x;
  key[t] = Rs[t * LSL + l];
  idx[t] = t;
  for (int k = 2; k <= 256; k <<= 1) {
    for (int j = k >> 1; j > 0; j >>= 1) {
      __syncthreads();
      const int p = t ^ j;
      if (p > t) {
        const bool up = ((t & k) == 0);
        const float a = key[t], c = key[p];
        if (up ? (a > c) : (a < c)) {
          key[t] = c; key[p] = a;
          const int ia = idx[t]; idx[t] = idx[p]; idx[p] = ia;
        }
      }
    }
  }
  __syncthreads();
  Rind[t * LSL + l] = idx[t];
}

// ---------------------------------------------------------------------------
// 5. Per (b,l): bitonic-sort 2048 in LDS, closed-form uniform-grid interp to
//    256 points, gather by Rind, subtract from Rslices, store.
// ---------------------------------------------------------------------------
__global__ void __launch_bounds__(256)
sort_interp_kernel(const float* __restrict__ XsT, const float* __restrict__ Rs,
                   const int* __restrict__ Rind, float* __restrict__ out) {
  __shared__ float vals[NPTS];
  __shared__ float itp[MREF];
  const int bl = blockIdx.x;          // b * 512 + l
  const int l  = bl & (LSL - 1);
  const int b  = bl >> 9;
  const int t  = threadIdx.x;

  const float* src = XsT + (size_t)bl * NPTS;
  for (int i = t; i < NPTS; i += 256) vals[i] = src[i];

  for (int k = 2; k <= NPTS; k <<= 1) {
    for (int j = k >> 1; j > 0; j >>= 1) {
      __syncthreads();
      for (int i = t; i < NPTS; i += 256) {
        const int p = i ^ j;
        if (p > i) {
          const bool up = ((i & k) == 0);
          const float a = vals[i], c = vals[p];
          if (up ? (a > c) : (a < c)) { vals[i] = c; vals[p] = a; }
        }
      }
    }
  }
  __syncthreads();

  // interp: xg[i]=(i+1)/2049, xn[m]=(m+1)/257; searchsorted in closed form
  {
    const int m = t;
    const float xn = (float)(m + 1) * (1.0f / (float)(MREF + 1));
    const int j = (int)ceilf(xn * (float)(NPTS + 1) - 1.0f);
    int idx = j - 1;
    if (idx < 0) idx = 0;
    if (idx > NPTS - 2) idx = NPTS - 2;
    const float invg = 1.0f / (float)(NPTS + 1);
    const float xg   = (float)(idx + 1) * invg;
    const float slope = (vals[idx + 1] - vals[idx]) / (1.1920928955078125e-7f + invg);
    itp[m] = vals[idx] + slope * (xn - xg);
  }
  __syncthreads();

  {
    const int m = t;
    const int r = Rind[m * LSL + l];
    out[((size_t)b * LSL + l) * MREF + m] = Rs[m * LSL + l] - itp[r];
  }
}

// ---------------------------------------------------------------------------
// launch
// ---------------------------------------------------------------------------
extern "C" void kernel_launch(void* const* d_in, const int* in_sizes, int n_in,
                              void* d_out, int out_size, void* d_ws, size_t ws_size,
                              hipStream_t stream) {
  const float* X     = (const float*)d_in[0];   // [32, 2048, 512]
  const float* theta = (const float*)d_in[1];   // [512, 512]
  const float* ref   = (const float*)d_in[2];   // [256, 512]
  float* out = (float*)d_out;                   // [32, 512*256]

  char* ws = (char*)d_ws;
  float* w    = (float*)ws;                                       // 1 MB
  float* XsT  = (float*)(ws + ((size_t)LSL * D_IN * 4));          // 128 MB  [B,L,N]
  float* Rs   = (float*)((char*)XsT + (size_t)BATCH * LSL * NPTS * 4);  // 0.5 MB
  int*   Rind = (int*)((char*)Rs + (size_t)MREF * LSL * 4);       // 0.5 MB

  normalize_rows_kernel<<<LSL, 256, 0, stream>>>(theta, w);

  dim3 grid(65536 / 128, LSL / 128);
  gemm_xw_kernel<<<grid, 256, 0, stream>>>(X, w, XsT);

  rslices_kernel<<<LSL, 256, 0, stream>>>(ref, w, Rs);
  argsort_kernel<<<LSL, 256, 0, stream>>>(Rs, Rind);
  sort_interp_kernel<<<BATCH * LSL, 256, 0, stream>>>(XsT, Rs, Rind, out);
}
